// Mamba1_46059229283151
// MI455X (gfx1250) — matrix operand order repaired and checked by reading, not compiled, not run on hardware
//
#include <hip/hip_runtime.h>
#include <hip/hip_bf16.h>
#include <math.h>

typedef __attribute__((ext_vector_type(16))) _Float16 v16h;
typedef __attribute__((ext_vector_type(8)))  float    v8f;

#define D_MODEL 1024
#define D_INNER 2048
#define DT_RANK 64
#define D_STATE 16
#define D_CONV  4
#define BATCH   2
#define SEQ     1024
#define NROWS   (BATCH * SEQ)            // 2048 token rows
#define XDBL_N  (DT_RANK + 2 * D_STATE)  // 96

// ---------------------------------------------------------------------------
// Fragment loaders (wave32 layouts, CDNA5 ISA §7.12.2):
//   A 16x32 f16 : lane m=l&15, hi=l>>4 -> K = k0+hi*8+[0..7] and k0+16+hi*8+[0..7]
//   B 32x16 f16 : lane n=l&15, hi=l>>4 -> K = k0+hi*8+[0..7] and k0+16+hi*8+[0..7] of W row n
// Source data is fp32 row-major with K contiguous; convert in registers
// (lowers to packed v_cvt_pk_f16_f32).
// ---------------------------------------------------------------------------
__device__ __forceinline__ v16h load_a_frag(const float* __restrict__ arow,
                                            int k0, int hi) {
    v16h f;
    const float* pa = arow + k0 + hi * 8;
#pragma unroll
    for (int i = 0; i < 8; ++i) f[i]     = (_Float16)pa[i];
#pragma unroll
    for (int i = 0; i < 8; ++i) f[8 + i] = (_Float16)pa[16 + i];
    return f;
}

__device__ __forceinline__ v16h load_b_frag(const float* __restrict__ brow,
                                            int k0, int hi) {
    v16h f;
    const float* pb = brow + k0 + hi * 8;
#pragma unroll
    for (int i = 0; i < 8; ++i) f[i]     = (_Float16)pb[i];
#pragma unroll
    for (int i = 0; i < 8; ++i) f[8 + i] = (_Float16)pb[16 + i];
    return f;
}

#define WMMA(af, bf, c) \
    __builtin_amdgcn_wmma_f32_16x16x32_f16(false, (af), false, (bf), (short)0, (c), false, false)

// ---------------------------------------------------------------------------
// WMMA GEMM: D(MxN) = A(MxK,row-major) * W(NxK,row-major)^T
// One wave computes a 32x32 tile (2 A-frags x 2 B-frags -> 4 v_wmma / K-step),
// 4 waves per block tiled along N.  mode 1: softplus(acc + bias[n]).
// ---------------------------------------------------------------------------
__global__ __launch_bounds__(128) void wmma_gemm_kernel(
    const float* __restrict__ A, int lda,
    const float* __restrict__ W, int ldb,
    float* __restrict__ D, int ldd,
    int M, int N, int K,
    const float* __restrict__ bias, int mode)
{
    const int lane  = threadIdx.x & 31;
    const int wave  = threadIdx.x >> 5;
    const int mbase = blockIdx.x * 32;
    const int nbase = (blockIdx.y * 4 + wave) * 32;
    if (nbase >= N) return;               // uniform per wave: EXEC stays all-1s

    const int mn = lane & 15;             // row-in-tile for A, col-in-tile for B/D
    const int hi = lane >> 4;

    const float* __restrict__ arow0 = A + (size_t)(mbase + mn) * lda;
    const float* __restrict__ arow1 = arow0 + (size_t)16 * lda;
    const float* __restrict__ brow0 = W + (size_t)(nbase + mn) * ldb;
    const float* __restrict__ brow1 = brow0 + (size_t)16 * ldb;

    v8f c00 = {}, c01 = {}, c10 = {}, c11 = {};
    for (int k0 = 0; k0 < K; k0 += 32) {
        if (k0 + 32 < K) {                // pull next K-slab toward the WGP
            __builtin_prefetch(arow0 + k0 + 32 + hi * 8,  0, 1);
            __builtin_prefetch(arow1 + k0 + 32 + hi * 8,  0, 1);
            __builtin_prefetch(brow0 + k0 + 32 + hi * 16, 0, 1);
            __builtin_prefetch(brow1 + k0 + 32 + hi * 16, 0, 1);
        }
        const v16h af0 = load_a_frag(arow0, k0, hi);
        const v16h af1 = load_a_frag(arow1, k0, hi);
        const v16h bf0 = load_b_frag(brow0, k0, hi);
        const v16h bf1 = load_b_frag(brow1, k0, hi);
        c00 = WMMA(af0, bf0, c00);
        c01 = WMMA(af0, bf1, c01);
        c10 = WMMA(af1, bf0, c10);
        c11 = WMMA(af1, bf1, c11);
    }

    // C/D layout: lane -> column n=l&15, rows (l>>4)*8 + [0..7]
    const int m0 = hi * 8;
    float bv0 = 0.f, bv1 = 0.f;
    if (mode == 1) { bv0 = bias[nbase + mn]; bv1 = bias[nbase + 16 + mn]; }

#pragma unroll
    for (int r = 0; r < 8; ++r) {
        float v00 = c00[r], v01 = c01[r], v10 = c10[r], v11 = c11[r];
        if (mode == 1) {   // softplus: max(x,0) + log(1+exp(-|x|)) -> v_exp+v_log
            v00 += bv0; v01 += bv1; v10 += bv0; v11 += bv1;
            v00 = fmaxf(v00, 0.f) + __logf(1.f + __expf(-fabsf(v00)));
            v01 = fmaxf(v01, 0.f) + __logf(1.f + __expf(-fabsf(v01)));
            v10 = fmaxf(v10, 0.f) + __logf(1.f + __expf(-fabsf(v10)));
            v11 = fmaxf(v11, 0.f) + __logf(1.f + __expf(-fabsf(v11)));
        }
        float* drow0 = D + (size_t)(mbase + m0 + r) * ldd + nbase + mn;
        float* drow1 = D + (size_t)(mbase + 16 + m0 + r) * ldd + nbase + mn;
        drow0[0]  = v00;
        drow0[16] = v01;
        drow1[0]  = v10;
        drow1[16] = v11;
    }
}

// ---------------------------------------------------------------------------
// Causal depthwise conv1d (K=4) + SiLU.  One thread per channel d with a
// register shift-history along l; loads are coalesced across d at each l.
// ---------------------------------------------------------------------------
__global__ __launch_bounds__(256) void conv_silu_kernel(
    const float* __restrict__ Y1,        // (NROWS, 2*D_INNER); cols [0,2048)=xi
    const float* __restrict__ conv_w,    // (D_INNER, 1, 4)
    const float* __restrict__ conv_b,    // (D_INNER,)
    float* __restrict__ xi_conv)         // (NROWS, D_INNER)
{
    const int d = blockIdx.x * blockDim.x + threadIdx.x;   // 0..D_INNER-1
    const int b = blockIdx.y;
    const float w0 = conv_w[d * 4 + 0], w1 = conv_w[d * 4 + 1];
    const float w2 = conv_w[d * 4 + 2], w3 = conv_w[d * 4 + 3];
    const float bs = conv_b[d];
    float x0 = 0.f, x1 = 0.f, x2 = 0.f;
    for (int l = 0; l < SEQ; ++l) {
        const size_t row = (size_t)b * SEQ + l;
        const float x3 = Y1[row * (2 * D_INNER) + d];
        float v = w0 * x0 + w1 * x1 + w2 * x2 + w3 * x3 + bs;
        v = v / (1.0f + __expf(-v));                       // SiLU
        xi_conv[row * D_INNER + d] = v;
        x0 = x1; x1 = x2; x2 = x3;
    }
}

// ---------------------------------------------------------------------------
// Selective scan + D skip + SiLU(res) gating.
// One thread per channel d keeps its 16 states in registers; B/C for the
// current (b,l) are staged through LDS (shared by all 256 channels in block).
// ---------------------------------------------------------------------------
__global__ __launch_bounds__(256) void scan_kernel(
    const float* __restrict__ xi_conv,   // (NROWS, D_INNER)  = u
    const float* __restrict__ x_dbl,     // (NROWS, 96): [0:64)=dt, [64:80)=B, [80:96)=C
    const float* __restrict__ delta,     // (NROWS, D_INNER)
    const float* __restrict__ A_log,     // (D_INNER, 16)
    const float* __restrict__ Dp,        // (D_INNER,)
    const float* __restrict__ Y1,        // res at cols [2048, 4096)
    float* __restrict__ ygated)          // (NROWS, D_INNER)
{
    const int d = blockIdx.x * blockDim.x + threadIdx.x;
    const int b = blockIdx.y;
    __shared__ float BC[2 * D_STATE];    // [0:16)=B, [16:32)=C

    float Arow[D_STATE], h[D_STATE];
#pragma unroll
    for (int n = 0; n < D_STATE; ++n) {
        Arow[n] = -__expf(A_log[(size_t)d * D_STATE + n]);
        h[n] = 0.0f;
    }
    const float Dd = Dp[d];

    for (int l = 0; l < SEQ; ++l) {
        const size_t row = (size_t)b * SEQ + l;
        if (threadIdx.x < 2 * D_STATE)
            BC[threadIdx.x] = x_dbl[row * XDBL_N + DT_RANK + threadIdx.x];
        __syncthreads();

        const float dlt = delta[row * D_INNER + d];
        const float u   = xi_conv[row * D_INNER + d];
        const float dbu = dlt * u;
        float sum = 0.0f;
#pragma unroll
        for (int n = 0; n < D_STATE; ++n) {
            const float dA = __expf(dlt * Arow[n]);
            h[n] = dA * h[n] + dbu * BC[n];
            sum += h[n] * BC[D_STATE + n];
        }
        float y = sum + u * Dd;
        const float r = Y1[row * (2 * D_INNER) + D_INNER + d];
        y *= r / (1.0f + __expf(-r));                      // * SiLU(res)
        ygated[row * D_INNER + d] = y;
        __syncthreads();                 // protect BC before next overwrite
    }
}

// ---------------------------------------------------------------------------
extern "C" void kernel_launch(void* const* d_in, const int* in_sizes, int n_in,
                              void* d_out, int out_size, void* d_ws, size_t ws_size,
                              hipStream_t stream) {
    const float* x          = (const float*)d_in[0];  // (B,L,1024)
    const float* in_proj_w  = (const float*)d_in[1];  // (4096,1024)
    const float* conv_w     = (const float*)d_in[2];  // (2048,1,4)
    const float* conv_b     = (const float*)d_in[3];  // (2048,)
    const float* x_proj_w   = (const float*)d_in[4];  // (96,2048)
    const float* dt_proj_w  = (const float*)d_in[5];  // (2048,64)
    const float* dt_proj_b  = (const float*)d_in[6];  // (2048,)
    const float* A_log      = (const float*)d_in[7];  // (2048,16)
    const float* Dp         = (const float*)d_in[8];  // (2048,)
    const float* out_proj_w = (const float*)d_in[9];  // (1024,2048)
    float* out = (float*)d_out;                       // (B,L,1024)

    // Workspace layout (floats)
    float* Y1      = (float*)d_ws;                          // 2048 x 4096
    float* xi_conv = Y1      + (size_t)NROWS * 2 * D_INNER; // 2048 x 2048
    float* x_dbl   = xi_conv + (size_t)NROWS * D_INNER;     // 2048 x 96
    float* delta   = x_dbl   + (size_t)NROWS * XDBL_N;      // 2048 x 2048
    float* yg      = delta   + (size_t)NROWS * D_INNER;     // 2048 x 2048

    const dim3 gblk(128);
    const int mgrid = NROWS / 32;        // 64

    // 1) in_proj: Y1 = x @ in_proj_w^T   (2048 x 4096, K=1024)
    wmma_gemm_kernel<<<dim3(mgrid, (2 * D_INNER) / 128), gblk, 0, stream>>>(
        x, D_MODEL, in_proj_w, D_MODEL, Y1, 2 * D_INNER,
        NROWS, 2 * D_INNER, D_MODEL, nullptr, 0);

    // 2) causal depthwise conv + SiLU on xi half of Y1
    conv_silu_kernel<<<dim3(D_INNER / 256, BATCH), 256, 0, stream>>>(
        Y1, conv_w, conv_b, xi_conv);

    // 3) x_proj: x_dbl = xi_conv @ x_proj_w^T   (2048 x 96, K=2048)
    wmma_gemm_kernel<<<dim3(mgrid, 1), gblk, 0, stream>>>(
        xi_conv, D_INNER, x_proj_w, D_INNER, x_dbl, XDBL_N,
        NROWS, XDBL_N, D_INNER, nullptr, 0);

    // 4) dt_proj + bias + softplus: delta (2048 x 2048, K=64); A = dt cols of x_dbl
    wmma_gemm_kernel<<<dim3(mgrid, D_INNER / 128), gblk, 0, stream>>>(
        x_dbl, XDBL_N, dt_proj_w, DT_RANK, delta, D_INNER,
        NROWS, D_INNER, DT_RANK, dt_proj_b, 1);

    // 5) selective scan + D skip + SiLU(res) gating
    scan_kernel<<<dim3(D_INNER / 256, BATCH), 256, 0, stream>>>(
        xi_conv, x_dbl, delta, A_log, Dp, Y1, yg);

    // 6) out_proj: out = yg @ out_proj_w^T   (2048 x 1024, K=2048)
    wmma_gemm_kernel<<<dim3(mgrid, D_MODEL / 128), gblk, 0, stream>>>(
        yg, D_INNER, out_proj_w, D_INNER, out, D_MODEL,
        NROWS, D_MODEL, D_INNER, nullptr, 0);
}
